// GCNN_2456721293449
// MI455X (gfx1250) — compile-verified, matrix operand order checked by reading, not run on hardware
//
#include <hip/hip_runtime.h>
#include <hip/hip_bf16.h>

typedef __attribute__((ext_vector_type(16))) __bf16 v16bf;
typedef __attribute__((ext_vector_type(8)))  float  v8f;

#define NNODES 50000
#define NEDGES 800000
#define NGRAPH 64
#define HDIM   128

// ---------------- utility kernels ----------------

__global__ void k_zero(float* __restrict__ p, int n) {
    int t = blockIdx.x * blockDim.x + threadIdx.x;
    if (t < n) p[t] = 0.0f;
}

__global__ void k_cvt_bf16(const float* __restrict__ s, __bf16* __restrict__ d, int n) {
    int t = blockIdx.x * blockDim.x + threadIdx.x;
    if (t < n) d[t] = (__bf16)s[t];
}

__global__ void k_count(const int* __restrict__ batch, int* __restrict__ cnt, int n) {
    int t = blockIdx.x * blockDim.x + threadIdx.x;
    if (t < n) atomicAdd(&cnt[batch[t]], 1);
}

// ---------------- edge scatter: agg[dst] += x[src] ----------------
// one thread = one edge x one 4-float chunk (32 chunks of 128 feats)
__global__ void k_scatter(const int* __restrict__ edge_index,
                          const float* __restrict__ xin,
                          float* __restrict__ agg, int E) {
    int t = blockIdx.x * blockDim.x + threadIdx.x;
    int e = t >> 5;
    int c = t & 31;
    if (e >= E) return;
    int s = edge_index[e];
    int d = edge_index[E + e];
    const float4 v = *(const float4*)(xin + (size_t)s * HDIM + c * 4);
    float* ap = agg + (size_t)d * HDIM + c * 4;
    atomicAdd(ap + 0, v.x);
    atomicAdd(ap + 1, v.y);
    atomicAdd(ap + 2, v.z);
    atomicAdd(ap + 3, v.w);
}

// ---------------- WMMA fragment loaders ----------------
// A-matrix 16x32 bf16 layout (wave32): lanes 0-15 hold M=0..15 with
// K = kb + kh*8 .. +7 (elems 0..7) and K = kb+16+kh*8 .. +7 (elems 8..15),
// kh = lane/16. B is the mirror with N in place of M, so a lane assigned to
// output column n just loads row n of W (since B[k][n] = W[n][k]).

__device__ inline v16bf frag_from_f32(const float* __restrict__ base,
                                      int row, int kb, int kh) {
    v16bf v;
    const float* p0 = base + (size_t)row * HDIM + kb + kh * 8;
    const float* p1 = p0 + 16;
#pragma unroll
    for (int j = 0; j < 8; ++j) {
        v[j]     = (__bf16)p0[j];
        v[8 + j] = (__bf16)p1[j];
    }
    return v;
}

__device__ inline v16bf frag_from_bf16(const __bf16* __restrict__ base,
                                       int row, int kb, int kh) {
    v16bf v;
    const __bf16* p0 = base + (size_t)row * HDIM + kb + kh * 8;
    const __bf16* p1 = p0 + 16;
#pragma unroll
    for (int j = 0; j < 8; ++j) {
        v[j]     = p0[j];
        v[8 + j] = p1[j];
    }
    return v;
}

// ---------------- fused GraphConv GEMM ----------------
// out = relu(agg @ Wrel^T + x @ Wroot^T + b), all [N,128] x [128,128]
// 256 threads = 8 waves; each wave owns a 16-row strip, loops over 8
// 16-column tiles; K=128 -> 4 bf16 WMMAs per operand, 8 per tile.
__global__ __launch_bounds__(256) void k_gemm(const float* __restrict__ agg,
                                              const float* __restrict__ xin,
                                              const __bf16* __restrict__ Wrel,
                                              const __bf16* __restrict__ Wroot,
                                              const float* __restrict__ bias,
                                              float* __restrict__ out, int Nrows) {
    int lane = threadIdx.x & 31;
    int wave = threadIdx.x >> 5;
    int mtile = blockIdx.x * 8 + wave;
    int row0 = mtile * 16;
    if (row0 >= Nrows) return;            // whole-wave exit: EXEC stays full for WMMA

    int mload = row0 + (lane & 15);
    if (mload >= Nrows) mload = Nrows - 1;
    int kh = lane >> 4;

    v16bf a_agg[4], a_x[4];
#pragma unroll
    for (int f = 0; f < 4; ++f) {
        a_agg[f] = frag_from_f32(agg, mload, f * 32, kh);
        a_x[f]   = frag_from_f32(xin, mload, f * 32, kh);
    }

#pragma unroll
    for (int ct = 0; ct < 8; ++ct) {
        int n = ct * 16 + (lane & 15);
        v8f c = {};
#pragma unroll
        for (int f = 0; f < 4; ++f) {
            v16bf b = frag_from_bf16(Wrel, n, f * 32, kh);
            c = __builtin_amdgcn_wmma_f32_16x16x32_bf16(false, a_agg[f], false, b,
                                                        (short)0, c, false, false);
        }
#pragma unroll
        for (int f = 0; f < 4; ++f) {
            v16bf b = frag_from_bf16(Wroot, n, f * 32, kh);
            c = __builtin_amdgcn_wmma_f32_16x16x32_bf16(false, a_x[f], false, b,
                                                        (short)0, c, false, false);
        }
        float bn = bias[n];
#pragma unroll
        for (int r = 0; r < 8; ++r) {
            int m = row0 + ((lane < 16) ? r : r + 8);
            if (m < Nrows) {
                float v = c[r] + bn;
                out[(size_t)m * HDIM + ct * 16 + (lane & 15)] = v > 0.0f ? v : 0.0f;
            }
        }
    }
}

// ---------------- pooling ----------------
// values are post-ReLU (>= 0), so uint-ordered atomicMax on zero-initialized
// buffers == segment_max with the reference's empty-segment->0 guard.
__global__ void k_pool(const float* __restrict__ h, const int* __restrict__ batch,
                       float* __restrict__ mx, float* __restrict__ sm, int Nrows) {
    int t = blockIdx.x * blockDim.x + threadIdx.x;
    int i = t >> 7;
    int f = t & 127;
    if (i >= Nrows) return;
    int g = batch[i];
    float v = h[(size_t)i * HDIM + f];
    atomicMax((unsigned int*)(mx + g * HDIM + f), __float_as_uint(v));
    atomicAdd(sm + g * HDIM + f, v);
}

__global__ void k_combine(const float* __restrict__ mx, const float* __restrict__ sm,
                          const int* __restrict__ cnt, float* __restrict__ gacc) {
    int t = blockIdx.x * blockDim.x + threadIdx.x;
    if (t >= NGRAPH * HDIM) return;
    int g = t >> 7;
    int f = t & 127;
    float c = (float)cnt[g];
    if (c < 1.0f) c = 1.0f;
    gacc[g * 256 + f]       += mx[t];
    gacc[g * 256 + 128 + f] += sm[t] / c;
}

// ---------------- tiny MLP head ----------------
__global__ void k_mlp(const float* __restrict__ in, const float* __restrict__ W,
                      const float* __restrict__ b, float* __restrict__ out,
                      int K, int O, int doRelu) {
    int t = blockIdx.x * blockDim.x + threadIdx.x;
    if (t >= NGRAPH * O) return;
    int g = t / O, o = t % O;
    float acc = b[o];
    const float* ip = in + g * K;
    const float* wp = W + o * K;
    for (int k = 0; k < K; ++k) acc += ip[k] * wp[k];
    if (doRelu && acc < 0.0f) acc = 0.0f;
    out[g * O + o] = acc;
}

// ---------------- host launcher ----------------

extern "C" void kernel_launch(void* const* d_in, const int* in_sizes, int n_in,
                              void* d_out, int out_size, void* d_ws, size_t ws_size,
                              hipStream_t stream) {
    (void)in_sizes; (void)n_in; (void)out_size; (void)ws_size;

    const float* x          = (const float*)d_in[0];
    const int*   edge_index = (const int*)d_in[1];
    const int*   batch      = (const int*)d_in[2];
    const float* W_rel[3]  = {(const float*)d_in[3], (const float*)d_in[6], (const float*)d_in[9]};
    const float* b_rel[3]  = {(const float*)d_in[4], (const float*)d_in[7], (const float*)d_in[10]};
    const float* W_root[3] = {(const float*)d_in[5], (const float*)d_in[8], (const float*)d_in[11]};
    const float* W1 = (const float*)d_in[12]; const float* b1 = (const float*)d_in[13];
    const float* W2 = (const float*)d_in[14]; const float* b2 = (const float*)d_in[15];
    const float* W3 = (const float*)d_in[16]; const float* b3 = (const float*)d_in[17];

    char* ws = (char*)d_ws;
    const size_t NH = (size_t)NNODES * HDIM;          // 6.4M floats
    float*  agg = (float*)(ws);                        // 25.6 MB
    float*  h1  = (float*)(ws + NH * 4);               // 25.6 MB
    float*  h2  = (float*)(ws + 2 * NH * 4);           // 25.6 MB
    __bf16* wbf = (__bf16*)(ws + 3 * NH * 4);          // 6 * 16384 bf16
    char*   p   = ws + 3 * NH * 4 + 6 * 16384 * sizeof(__bf16);
    float*  mx  = (float*)p;                p += NGRAPH * HDIM * 4;
    float*  sm  = (float*)p;                p += NGRAPH * HDIM * 4;
    int*    cnt = (int*)p;                  p += 256;
    float*  z1  = (float*)p;                p += NGRAPH * 128 * 4;
    float*  z2  = (float*)p;

    float* logits    = (float*)d_out;                       // [64,10]
    float* node_embs = (float*)d_out + 640;                 // [N,128] = layer-3 output
    float* graph_emb = (float*)d_out + 640 + NH;            // [64,256] accumulator

    const int T = 256;
    // convert weights to bf16 (deterministic, every call)
    for (int l = 0; l < 3; ++l) {
        k_cvt_bf16<<<(16384 + T - 1) / T, T, 0, stream>>>(W_rel[l],  wbf + (2 * l) * 16384,     16384);
        k_cvt_bf16<<<(16384 + T - 1) / T, T, 0, stream>>>(W_root[l], wbf + (2 * l + 1) * 16384, 16384);
    }
    // per-graph node counts
    k_zero<<<1, T, 0, stream>>>((float*)cnt, 64);
    k_count<<<(NNODES + T - 1) / T, T, 0, stream>>>(batch, cnt, NNODES);
    // graph_emb accumulator = 0
    k_zero<<<(NGRAPH * 256 + T - 1) / T, T, 0, stream>>>(graph_emb, NGRAPH * 256);

    const float* layer_in[3] = {x, h1, h2};
    float*       layer_out[3] = {h1, h2, node_embs};
    const int nh_blocks   = (int)((NH + T - 1) / T);
    const int sc_blocks   = (int)(((size_t)NEDGES * 32 + T - 1) / T);
    const int gemm_blocks = ((NNODES + 15) / 16 + 7) / 8;   // 3125 tiles / 8 waves
    const int pool_blocks = (int)((NH + T - 1) / T);

    for (int l = 0; l < 3; ++l) {
        k_zero<<<nh_blocks, T, 0, stream>>>(agg, (int)NH);
        k_scatter<<<sc_blocks, T, 0, stream>>>(edge_index, layer_in[l], agg, NEDGES);
        k_gemm<<<gemm_blocks, T, 0, stream>>>(agg, layer_in[l],
                                              wbf + (2 * l) * 16384, wbf + (2 * l + 1) * 16384,
                                              b_rel[l], layer_out[l], NNODES);
        k_zero<<<(NGRAPH * HDIM + T - 1) / T, T, 0, stream>>>(mx, NGRAPH * HDIM);
        k_zero<<<(NGRAPH * HDIM + T - 1) / T, T, 0, stream>>>(sm, NGRAPH * HDIM);
        k_pool<<<pool_blocks, T, 0, stream>>>(layer_out[l], batch, mx, sm, NNODES);
        k_combine<<<(NGRAPH * HDIM + T - 1) / T, T, 0, stream>>>(mx, sm, cnt, graph_emb);
    }

    // MLP head: [64,256]->[64,128]->[64,64]->[64,10]
    k_mlp<<<(NGRAPH * 128 + T - 1) / T, T, 0, stream>>>(graph_emb, W1, b1, z1, 256, 128, 1);
    k_mlp<<<(NGRAPH * 64  + T - 1) / T, T, 0, stream>>>(z1,        W2, b2, z2, 128, 64, 1);
    k_mlp<<<(NGRAPH * 10  + T - 1) / T, T, 0, stream>>>(z2,        W3, b3, logits, 64, 10, 0);
}
